// MoEModel_51247549776485
// MI455X (gfx1250) — matrix operand order repaired and checked by reading, not compiled, not run on hardware
//
#include <hip/hip_runtime.h>
#include <hip/hip_bf16.h>

// ---------------------------------------------------------------------------
// MoE transformer forward for MI455X (gfx1250), wave32 + WMMA bf16.
// Heavy math via v_wmma_f32_16x16x32_bf16; fp32 -> bf16 conversion happens
// once per tile during vectorized (b128) LDS staging, branch-free.
// Block tile 128x128 (8 waves, 32x64 per wave), K-stage 64 (2 WMMA steps).
// ---------------------------------------------------------------------------

typedef __bf16 bf16_t;
typedef __attribute__((ext_vector_type(16))) __bf16 v16bf;
typedef __attribute__((ext_vector_type(8)))  float  v8f;

// Model constants (from the reference)
#define MV    32000
#define MC    1024
#define MNH   16
#define MNKV  4
#define ML    4
#define MND   1
#define MT    1024
#define ME    8
#define MHD   64
#define MHID  2736
#define MKVD  (MNKV * MHD)   // 256

#define BM 128
#define BN 128
#define BK 64

#define GF_ACCUM    1
#define GF_ROWSCALE 2

// ---------------------------------------------------------------------------
// bf16 packing helper (lowers to v_cvt_pk_bf16_f32)
// ---------------------------------------------------------------------------
__device__ inline unsigned pack_bf16x2(float a, float b) {
  union { bf16_t h[2]; unsigned u; } r;
  r.h[0] = (bf16_t)a; r.h[1] = (bf16_t)b;
  return r.u;
}

// ---------------------------------------------------------------------------
// Fragment loaders per CDNA5 ISA 16-bit layouts (CDNA5 ISA §7.12.2).
// A (16x32 MxK) at K-window kk: lane l -> row M=l&15; half=l>>4;
//   VGPR j(0..3): K=kk+half*8+2j, VGPR j(4..7): K=kk+16+half*8+2(j-4).
// ---------------------------------------------------------------------------
__device__ inline v16bf load_a_frag(const bf16_t (&As)[BM][BK], int mbase, int kk, int lane) {
  union { v16bf v; unsigned u[8]; } r;
  const int m  = mbase + (lane & 15);
  const int w0 = (kk >> 1) + ((lane >> 4) << 2);   // word index of K=kk+half*8
  const unsigned* row = (const unsigned*)&As[m][0];
#pragma unroll
  for (int j = 0; j < 4; ++j) r.u[j]     = row[w0 + j];
#pragma unroll
  for (int j = 0; j < 4; ++j) r.u[4 + j] = row[w0 + 8 + j];
  return r.v;
}

// B (32x16 KxN) at K-window kk, Bs stored transposed [N][K]:
// lane l -> col N=l&15; half=l>>4; VGPR j(0..3): K=kk+half*8+2j, VGPR j(4..7): K=kk+16+half*8+2(j-4).
__device__ inline v16bf load_b_frag(const bf16_t (&Bs)[BN][BK], int nbase, int kk, int lane) {
  union { v16bf v; unsigned u[8]; } r;
  const int n  = nbase + (lane & 15);
  const int w0 = (kk >> 1) + ((lane >> 4) << 2);
  const unsigned* row = (const unsigned*)&Bs[n][0];
#pragma unroll
  for (int j = 0; j < 4; ++j) r.u[j]     = row[w0 + j];
#pragma unroll
  for (int j = 0; j < 4; ++j) r.u[4 + j] = row[w0 + 8 + j];
  return r.v;
}

// ---------------------------------------------------------------------------
// Generic GEMM: D[MxN] (+)= alpha * rowScale[m] * (A[MxK] @ B[KxN])
// A row-major (lda); B row-major KxN (ldb) or, if transB, row-major NxK.
// Requires K % 4 == 0 and N % 4 == 0 (true for every call in this model);
// K tails vs BK and N tails vs BN are zero-filled with predicated loads.
// ---------------------------------------------------------------------------
__global__ __launch_bounds__(256) void gemm_bf16_kernel(
    const float* __restrict__ A, const float* __restrict__ B,
    float* __restrict__ D,
    int M, int N, int Kdim, int lda, int ldb, int ldd,
    int transB, int flags, float alpha,
    const float* __restrict__ rowScale, int rsStride)
{
  __shared__ bf16_t As[BM][BK];
  __shared__ bf16_t Bs[BN][BK];

  const int tid  = threadIdx.x;
  const int lane = tid & 31;
  const int wave = tid >> 5;          // 0..7
  const int wm   = wave >> 1;         // 0..3 -> 32-row slice
  const int wn   = wave & 1;          // 0..1 -> 64-col slice
  const int bM   = blockIdx.y * BM;
  const int bN   = blockIdx.x * BN;

  v8f acc[2][4] = {};

  for (int k0 = 0; k0 < Kdim; k0 += BK) {
    // ---------------- stage A panel (BM x BK): 2048 float4 groups ----------
#pragma unroll
    for (int batch = 0; batch < 2; ++batch) {
      float4 av[4];
      int am[4], ak[4];
#pragma unroll
      for (int it = 0; it < 4; ++it) {
        const int e = (batch * 4 + it) * 256 + tid;
        am[it] = e >> 4;               // / (BK/4)
        ak[it] = (e & 15) << 2;
        const int gm = bM + am[it];
        const int gk = k0 + ak[it];
        const bool p = (gm < M) && (gk < Kdim);
        const int rm = (gm < M) ? gm : (M - 1);
        const int rk = p ? gk : 0;
        float4 v = *(const float4*)&A[(size_t)rm * lda + rk];
        av[it].x = p ? v.x : 0.f; av[it].y = p ? v.y : 0.f;
        av[it].z = p ? v.z : 0.f; av[it].w = p ? v.w : 0.f;
      }
#pragma unroll
      for (int it = 0; it < 4; ++it) {
        uint2 pk; pk.x = pack_bf16x2(av[it].x, av[it].y);
        pk.y = pack_bf16x2(av[it].z, av[it].w);
        *(uint2*)&As[am[it]][ak[it]] = pk;
      }
    }

    // ---------------- stage B panel (BK x BN), stored [N][K] ----------------
    if (transB) {
      // B is N x K (row-major): K-contiguous, same pattern as A.
#pragma unroll
      for (int batch = 0; batch < 2; ++batch) {
        float4 bv[4];
        int bn_[4], bk_[4];
#pragma unroll
        for (int it = 0; it < 4; ++it) {
          const int e = (batch * 4 + it) * 256 + tid;
          bn_[it] = e >> 4;
          bk_[it] = (e & 15) << 2;
          const int gn = bN + bn_[it];
          const int gk = k0 + bk_[it];
          const bool p = (gn < N) && (gk < Kdim);
          const int rn = (gn < N) ? gn : (N - 1);
          const int rk = p ? gk : 0;
          float4 v = *(const float4*)&B[(size_t)rn * ldb + rk];
          bv[it].x = p ? v.x : 0.f; bv[it].y = p ? v.y : 0.f;
          bv[it].z = p ? v.z : 0.f; bv[it].w = p ? v.w : 0.f;
        }
#pragma unroll
        for (int it = 0; it < 4; ++it) {
          uint2 pk; pk.x = pack_bf16x2(bv[it].x, bv[it].y);
          pk.y = pack_bf16x2(bv[it].z, bv[it].w);
          *(uint2*)&Bs[bn_[it]][bk_[it]] = pk;
        }
      }
    } else {
      // B is K x N (row-major): load N-contiguous float4, scatter-transpose.
#pragma unroll
      for (int batch = 0; batch < 2; ++batch) {
        float4 bv[4];
        int bn_[4], bk_[4];
#pragma unroll
        for (int it = 0; it < 4; ++it) {
          const int e = (batch * 4 + it) * 256 + tid;
          bk_[it] = e >> 5;            // / (BN/4)
          bn_[it] = (e & 31) << 2;
          const int gk = k0 + bk_[it];
          const int gn = bN + bn_[it];
          const bool p = (gk < Kdim) && (gn < N);
          const int rk = (gk < Kdim) ? gk : (Kdim - 1);
          const int rn = p ? gn : 0;
          float4 v = *(const float4*)&B[(size_t)rk * ldb + rn];
          bv[it].x = p ? v.x : 0.f; bv[it].y = p ? v.y : 0.f;
          bv[it].z = p ? v.z : 0.f; bv[it].w = p ? v.w : 0.f;
        }
#pragma unroll
        for (int it = 0; it < 4; ++it) {
          Bs[bn_[it] + 0][bk_[it]] = (bf16_t)bv[it].x;
          Bs[bn_[it] + 1][bk_[it]] = (bf16_t)bv[it].y;
          Bs[bn_[it] + 2][bk_[it]] = (bf16_t)bv[it].z;
          Bs[bn_[it] + 3][bk_[it]] = (bf16_t)bv[it].w;
        }
      }
    }

    // Prefetch this thread's next K-tile addresses (global_prefetch_b8).
    if (k0 + BK < Kdim) {
      const int gm = bM + (tid >> 4);
      const int rm = (gm < M) ? gm : (M - 1);
      const int gk = k0 + BK + ((tid & 15) << 2);
      __builtin_prefetch(&A[(size_t)rm * lda + ((gk < Kdim) ? gk : 0)], 0, 1);
      if (transB) {
        const int gn = bN + (tid >> 4);
        const int rn = (gn < N) ? gn : (N - 1);
        __builtin_prefetch(&B[(size_t)rn * ldb + ((gk < Kdim) ? gk : 0)], 0, 1);
      } else {
        const int gk2 = k0 + BK + (tid >> 5);
        const int rk = (gk2 < Kdim) ? gk2 : (Kdim - 1);
        const int gn = bN + ((tid & 31) << 2);
        __builtin_prefetch(&B[(size_t)rk * ldb + ((gn < N) ? gn : 0)], 0, 1);
      }
    }

    __syncthreads();

#pragma unroll
    for (int kk = 0; kk < BK; kk += 32) {
      v16bf af0 = load_a_frag(As, wm * 32 + 0,  kk, lane);
      v16bf af1 = load_a_frag(As, wm * 32 + 16, kk, lane);
      v16bf bf0 = load_b_frag(Bs, wn * 64 + 0,  kk, lane);
      v16bf bf1 = load_b_frag(Bs, wn * 64 + 16, kk, lane);
      v16bf bf2 = load_b_frag(Bs, wn * 64 + 32, kk, lane);
      v16bf bf3 = load_b_frag(Bs, wn * 64 + 48, kk, lane);

      acc[0][0] = __builtin_amdgcn_wmma_f32_16x16x32_bf16(false, af0, false, bf0, (short)0, acc[0][0], false, false);
      acc[0][1] = __builtin_amdgcn_wmma_f32_16x16x32_bf16(false, af0, false, bf1, (short)0, acc[0][1], false, false);
      acc[0][2] = __builtin_amdgcn_wmma_f32_16x16x32_bf16(false, af0, false, bf2, (short)0, acc[0][2], false, false);
      acc[0][3] = __builtin_amdgcn_wmma_f32_16x16x32_bf16(false, af0, false, bf3, (short)0, acc[0][3], false, false);
      acc[1][0] = __builtin_amdgcn_wmma_f32_16x16x32_bf16(false, af1, false, bf0, (short)0, acc[1][0], false, false);
      acc[1][1] = __builtin_amdgcn_wmma_f32_16x16x32_bf16(false, af1, false, bf1, (short)0, acc[1][1], false, false);
      acc[1][2] = __builtin_amdgcn_wmma_f32_16x16x32_bf16(false, af1, false, bf2, (short)0, acc[1][2], false, false);
      acc[1][3] = __builtin_amdgcn_wmma_f32_16x16x32_bf16(false, af1, false, bf3, (short)0, acc[1][3], false, false);
    }

    __syncthreads();
  }

  // --- epilogue: C/D layout: VGPR r -> M = r + 8*(lane>>4), N = lane&15 ---
  const int nlo  = lane & 15;
  const int half = lane >> 4;
#pragma unroll
  for (int mt = 0; mt < 2; ++mt)
#pragma unroll
    for (int nt = 0; nt < 4; ++nt)
#pragma unroll
      for (int r = 0; r < 8; ++r) {
        const int m = bM + wm * 32 + mt * 16 + r + half * 8;
        const int n = bN + wn * 64 + nt * 16 + nlo;
        if (m < M && n < N) {
          float v = acc[mt][nt][r] * alpha;
          if (flags & GF_ROWSCALE) v *= rowScale[(size_t)m * rsStride];
          if (flags & GF_ACCUM) D[(size_t)m * ldd + n] += v;
          else                  D[(size_t)m * ldd + n] = v;
        }
      }
}

// ---------------------------------------------------------------------------
// Elementwise / reduction helpers
// ---------------------------------------------------------------------------
__global__ __launch_bounds__(256) void embed_kernel(
    const int* __restrict__ idx, const float* __restrict__ wte,
    float* __restrict__ x)
{
  const int t   = blockIdx.x;
  const int tok = idx[t];
  for (int c = threadIdx.x; c < MC; c += 256)
    x[(size_t)t * MC + c] = wte[(size_t)tok * MC + c];
}

__global__ __launch_bounds__(256) void rmsnorm_kernel(
    const float* __restrict__ x, const float* __restrict__ w,
    float* __restrict__ o, int Cdim)
{
  const float* xr = x + (size_t)blockIdx.x * Cdim;
  float* orow     = o + (size_t)blockIdx.x * Cdim;
  __shared__ float red[8];
  const int tid = threadIdx.x, lane = tid & 31, wv = tid >> 5;
  float ss = 0.f;
  for (int c = tid; c < Cdim; c += 256) { const float v = xr[c]; ss += v * v; }
#pragma unroll
  for (int o2 = 16; o2 > 0; o2 >>= 1) ss += __shfl_down(ss, o2, 32);
  if (lane == 0) red[wv] = ss;
  __syncthreads();
  if (tid == 0) { float t2 = 0.f; for (int i = 0; i < 8; ++i) t2 += red[i]; red[0] = t2; }
  __syncthreads();
  const float r = rsqrtf(red[0] / (float)Cdim + 1e-6f);
  for (int c = tid; c < Cdim; c += 256) orow[c] = xr[c] * r * w[c];
}

__global__ __launch_bounds__(256) void rope_kernel(
    float* __restrict__ q, float* __restrict__ k)
{
  const int half  = MHD / 2;
  const int per_t = (MNH + MNKV) * half;
  const int i = blockIdx.x * 256 + threadIdx.x;
  if (i >= MT * per_t) return;
  const int t    = i / per_t;
  const int rem  = i % per_t;
  const int head = rem / half;
  const int d    = rem % half;
  const float inv = __powf(10000.f, -2.f * (float)d / (float)MHD);
  const float ang = (float)t * inv;
  const float c = __cosf(ang), s = __sinf(ang);
  float* base = (head < MNH)
      ? (q + ((size_t)t * MNH + head) * MHD)
      : (k + ((size_t)t * MNKV + (head - MNH)) * MHD);
  const float x1 = base[d], x2 = base[d + half];
  base[d]        = x1 * c - x2 * s;
  base[d + half] = x1 * s + x2 * c;
}

__global__ __launch_bounds__(256) void causal_softmax_kernel(float* __restrict__ s)
{
  const int row = blockIdx.x & (MT - 1);
  float* sr = s + (size_t)blockIdx.x * MT;
  __shared__ float red[8];
  const int tid = threadIdx.x, lane = tid & 31, wv = tid >> 5;

  float mx = -3.0e38f;
  for (int j = tid; j < MT; j += 256) {
    const float v = (j <= row) ? sr[j] : -3.0e38f;
    mx = fmaxf(mx, v);
  }
#pragma unroll
  for (int o2 = 16; o2 > 0; o2 >>= 1) mx = fmaxf(mx, __shfl_down(mx, o2, 32));
  if (lane == 0) red[wv] = mx;
  __syncthreads();
  if (tid == 0) { float m = red[0]; for (int i = 1; i < 8; ++i) m = fmaxf(m, red[i]); red[0] = m; }
  __syncthreads();
  mx = red[0];
  __syncthreads();

  float sum = 0.f;
  for (int j = tid; j < MT; j += 256) {
    const float e = (j <= row) ? __expf(sr[j] - mx) : 0.f;
    sr[j] = e; sum += e;
  }
#pragma unroll
  for (int o2 = 16; o2 > 0; o2 >>= 1) sum += __shfl_down(sum, o2, 32);
  if (lane == 0) red[wv] = sum;
  __syncthreads();
  if (tid == 0) { float t2 = 0.f; for (int i = 0; i < 8; ++i) t2 += red[i]; red[0] = t2; }
  __syncthreads();
  const float inv = 1.f / red[0];
  for (int j = tid; j < MT; j += 256) sr[j] *= inv;
}

__global__ __launch_bounds__(256) void silu_mul_kernel(
    const float* __restrict__ g, const float* __restrict__ u,
    float* __restrict__ o, int n)
{
  const int i = blockIdx.x * 256 + threadIdx.x;
  if (i < n) {
    const float gv = g[i];
    o[i] = gv / (1.f + __expf(-gv)) * u[i];
  }
}

// Top-2 gating: logits = h @ gw (C x E), softmax over top-2 -> cw (T x E).
__global__ __launch_bounds__(256) void gate_topk_kernel(
    const float* __restrict__ h, const float* __restrict__ gw,
    float* __restrict__ cw)
{
  const int token = blockIdx.x * 8 + (threadIdx.x >> 5);
  const int lane  = threadIdx.x & 31;
  const float* xr = h + (size_t)token * MC;
  float acc[ME];
#pragma unroll
  for (int e = 0; e < ME; ++e) acc[e] = 0.f;
  for (int c = lane; c < MC; c += 32) {
    const float xv = xr[c];
#pragma unroll
    for (int e = 0; e < ME; ++e) acc[e] += xv * gw[(size_t)c * ME + e];
  }
#pragma unroll
  for (int e = 0; e < ME; ++e)
#pragma unroll
    for (int o2 = 16; o2 > 0; o2 >>= 1) acc[e] += __shfl_down(acc[e], o2, 32);
  if (lane == 0) {
    int i0 = 0; float v0 = acc[0];
#pragma unroll
    for (int e = 1; e < ME; ++e) if (acc[e] > v0) { v0 = acc[e]; i0 = e; }
    int i1 = -1; float v1 = -3.0e38f;
#pragma unroll
    for (int e = 0; e < ME; ++e) if (e != i0 && acc[e] > v1) { v1 = acc[e]; i1 = e; }
    const float p0 = 1.f / (1.f + __expf(v1 - v0));
#pragma unroll
    for (int e = 0; e < ME; ++e) cw[(size_t)token * ME + e] = 0.f;
    cw[(size_t)token * ME + i0] = p0;
    cw[(size_t)token * ME + i1] = 1.f - p0;
  }
}

// ---------------------------------------------------------------------------
// Host orchestration
// ---------------------------------------------------------------------------
static inline void gemm(hipStream_t st, const float* A, const float* B, float* D,
                        int M, int N, int K, int lda, int ldb, int ldd,
                        int transB, int flags, float alpha,
                        const float* rs, int rsStride)
{
  dim3 grid((N + BN - 1) / BN, (M + BM - 1) / BM);
  gemm_bf16_kernel<<<grid, 256, 0, st>>>(A, B, D, M, N, K, lda, ldb, ldd,
                                         transB, flags, alpha, rs, rsStride);
}

extern "C" void kernel_launch(void* const* d_in, const int* in_sizes, int n_in,
                              void* d_out, int out_size, void* d_ws, size_t ws_size,
                              hipStream_t stream)
{
  (void)in_sizes; (void)n_in; (void)out_size; (void)ws_size;

  const int*   idx      = (const int*)  d_in[0];
  const float* wte      = (const float*)d_in[1];
  const float* Wq       = (const float*)d_in[2];
  const float* Wk       = (const float*)d_in[3];
  const float* Wv       = (const float*)d_in[4];
  const float* Wo       = (const float*)d_in[5];
  const float* ln1      = (const float*)d_in[6];
  const float* ln2      = (const float*)d_in[7];
  const float* dense_wg = (const float*)d_in[8];
  const float* dense_wu = (const float*)d_in[9];
  const float* dense_wd = (const float*)d_in[10];
  const float* gate_w   = (const float*)d_in[11];
  const float* exp_wg   = (const float*)d_in[12];
  const float* exp_wu   = (const float*)d_in[13];
  const float* exp_wd   = (const float*)d_in[14];
  const float* sh_wg    = (const float*)d_in[15];
  const float* sh_wu    = (const float*)d_in[16];
  const float* sh_wd    = (const float*)d_in[17];
  const float* lnf      = (const float*)d_in[18];
  const float* lm_head  = (const float*)d_in[19];
  float* out = (float*)d_out;

  // Workspace carve-out (256B aligned)
  char* ws = (char*)d_ws;
  size_t off = 0;
  auto carve = [&](size_t nfloats) -> float* {
    float* p = (float*)(ws + off);
    off += ((nfloats * sizeof(float)) + 255) & ~(size_t)255;
    return p;
  };
  float* x      = carve((size_t)MT * MC);
  float* h      = carve((size_t)MT * MC);
  float* q      = carve((size_t)MT * MC);
  float* kbuf   = carve((size_t)MT * MKVD);
  float* vbuf   = carve((size_t)MT * MKVD);
  float* scores = carve((size_t)MNH * MT * MT);
  float* attno  = carve((size_t)MT * MC);
  float* g      = carve((size_t)MT * MHID);
  float* u      = carve((size_t)MT * MHID);
  float* hm     = carve((size_t)MT * MHID);
  float* cw     = carve((size_t)MT * ME);

  const float attn_scale = 0.125f;  // 1/sqrt(HD=64)
  const int rep = MNH / MNKV;

  embed_kernel<<<MT, 256, 0, stream>>>(idx, wte, x);

  for (int l = 0; l < ML; ++l) {
    // ---- attention ----
    rmsnorm_kernel<<<MT, 256, 0, stream>>>(x, ln1 + (size_t)l * MC, h, MC);
    gemm(stream, h, Wq + (size_t)l * MC * MC,   q,    MT, MC,   MC, MC, MC,   MC,   0, 0, 1.f, nullptr, 0);
    gemm(stream, h, Wk + (size_t)l * MC * MKVD, kbuf, MT, MKVD, MC, MC, MKVD, MKVD, 0, 0, 1.f, nullptr, 0);
    gemm(stream, h, Wv + (size_t)l * MC * MKVD, vbuf, MT, MKVD, MC, MC, MKVD, MKVD, 0, 0, 1.f, nullptr, 0);

    {
      const int total = MT * (MNH + MNKV) * (MHD / 2);
      rope_kernel<<<(total + 255) / 256, 256, 0, stream>>>(q, kbuf);
    }

    for (int hh = 0; hh < MNH; ++hh) {
      const int kvh = hh / rep;
      // scores_h = (q_h @ k_h^T) * scale   [T x T]
      gemm(stream, q + (size_t)hh * MHD, kbuf + (size_t)kvh * MHD,
           scores + (size_t)hh * MT * MT,
           MT, MT, MHD, MC, MKVD, MT, /*transB=*/1, 0, attn_scale, nullptr, 0);
    }
    causal_softmax_kernel<<<MNH * MT, 256, 0, stream>>>(scores);
    for (int hh = 0; hh < MNH; ++hh) {
      const int kvh = hh / rep;
      // o_h = P_h @ v_h   [T x HD], written into (T, C) at column hh*HD
      gemm(stream, scores + (size_t)hh * MT * MT, vbuf + (size_t)kvh * MHD,
           attno + (size_t)hh * MHD,
           MT, MHD, MT, MT, MKVD, MC, 0, 0, 1.f, nullptr, 0);
    }
    // x += o @ Wo
    gemm(stream, attno, Wo + (size_t)l * MC * MC, x,
         MT, MC, MC, MC, MC, MC, 0, GF_ACCUM, 1.f, nullptr, 0);

    // ---- MLP ----
    rmsnorm_kernel<<<MT, 256, 0, stream>>>(x, ln2 + (size_t)l * MC, h, MC);
    if (l < MND) {
      gemm(stream, h, dense_wg, g, MT, MHID, MC, MC, MHID, MHID, 0, 0, 1.f, nullptr, 0);
      gemm(stream, h, dense_wu, u, MT, MHID, MC, MC, MHID, MHID, 0, 0, 1.f, nullptr, 0);
      silu_mul_kernel<<<(MT * MHID + 255) / 256, 256, 0, stream>>>(g, u, hm, MT * MHID);
      gemm(stream, hm, dense_wd, x, MT, MC, MHID, MHID, MC, MC, 0, GF_ACCUM, 1.f, nullptr, 0);
    } else {
      const int j = l - MND;
      gate_topk_kernel<<<MT / 8, 256, 0, stream>>>(h, gate_w + (size_t)j * MC * ME, cw);
      for (int e = 0; e < ME; ++e) {
        const float* wg = exp_wg + ((size_t)j * ME + e) * MC * MHID;
        const float* wu = exp_wu + ((size_t)j * ME + e) * MC * MHID;
        const float* wd = exp_wd + ((size_t)j * ME + e) * MHID * MC;
        gemm(stream, h, wg, g, MT, MHID, MC, MC, MHID, MHID, 0, 0, 1.f, nullptr, 0);
        gemm(stream, h, wu, u, MT, MHID, MC, MC, MHID, MHID, 0, 0, 1.f, nullptr, 0);
        silu_mul_kernel<<<(MT * MHID + 255) / 256, 256, 0, stream>>>(g, u, hm, MT * MHID);
        // x += cw[:,e] * (hm @ wd)
        gemm(stream, hm, wd, x, MT, MC, MHID, MHID, MC, MC, 0,
             GF_ACCUM | GF_ROWSCALE, 1.f, cw + e, ME);
      }
      // shared expert
      gemm(stream, h, sh_wg + (size_t)j * MC * MHID, g, MT, MHID, MC, MC, MHID, MHID, 0, 0, 1.f, nullptr, 0);
      gemm(stream, h, sh_wu + (size_t)j * MC * MHID, u, MT, MHID, MC, MC, MHID, MHID, 0, 0, 1.f, nullptr, 0);
      silu_mul_kernel<<<(MT * MHID + 255) / 256, 256, 0, stream>>>(g, u, hm, MT * MHID);
      gemm(stream, hm, sh_wd + (size_t)j * MHID * MC, x, MT, MC, MHID, MHID, MC, MC, 0, GF_ACCUM, 1.f, nullptr, 0);
    }
  }

  // final norm + lm_head
  rmsnorm_kernel<<<MT, 256, 0, stream>>>(x, lnf, h, MC);
  gemm(stream, h, lm_head, out, MT, MV, MC, MC, MV, MV, 0, 0, 1.f, nullptr, 0);
}
